// MultiLoRALayerMasking_44933947850968
// MI455X (gfx1250) — matrix operand order repaired and checked by reading, not compiled, not run on hardware
//
#include <hip/hip_runtime.h>

typedef __attribute__((ext_vector_type(16))) _Float16 v16h;
typedef __attribute__((ext_vector_type(8)))  float    v8f;
typedef __attribute__((ext_vector_type(4)))  float    f32x4;

#define B_SZ   8
#define SEQ    2048
#define D_IN   4096
#define D_OUT  4096
#define RANK   16
#define SCALING 1.0f   // alpha / rank = 16/16

#define WAVES       8
#define THREADS     (WAVES * 32)        // 256
#define KCHUNK      (D_IN / WAVES)      // 512 K per wave in stage 1
#define KITER       (KCHUNK / 32)       // 16 WMMAs per wave in stage 1
#define NTILES      (D_OUT / 16)        // 256 output N-tiles
#define NT_PER_WAVE (NTILES / WAVES)    // 32 N-tiles per wave in stage 2

__global__ __launch_bounds__(THREADS)
void multilora_fused_kernel(const float* __restrict__ x,
                            const float* __restrict__ A,
                            const float* __restrict__ Bm,
                            const int*   __restrict__ ids,
                            float* __restrict__ out)
{
    __shared__ float hpart[WAVES][16 * 16];  // per-wave partial h tiles
    __shared__ float hfinal[16 * 16];        // reduced h tile (f32)

    const int tid   = threadIdx.x;
    const int wave  = tid >> 5;
    const int lane  = tid & 31;
    const int mrow  = lane & 15;     // matrix row (A-frag) / column (B-frag)
    const int khalf = lane >> 4;     // which K half this lane holds

    const int bid   = blockIdx.x;
    const int b     = bid >> 7;              // / (SEQ/16)
    const int s0    = (bid & 127) * 16;      // sequence tile start
    const int a     = ids[b];

    // ---------------- Stage 1: h = x_tile(16 x D_IN) @ Bm[a]^T ----------------
    // A-frag lane layout (f16 16x32): row = mrow; VGPR0-3 <-> K[kb, kb+8),
    // VGPR4-7 <-> K[kb+16, kb+24), kb = 8*khalf.
    // B-frag lane layout (f16 32x16): col = mrow; same K order as A-frag.
    v8f c = {0.f, 0.f, 0.f, 0.f, 0.f, 0.f, 0.f, 0.f};

    const float* xrow = x  + (size_t)(b * SEQ + s0 + mrow) * D_IN + khalf * 8;
    const float* brow = Bm + ((size_t)a * RANK + mrow) * D_IN + khalf * 8;

    for (int it = 0; it < KITER; ++it) {
        const int k0 = wave * KCHUNK + it * 32;

        // x is streamed (read exactly once) -> non-temporal loads keep the
        // hot adapter weights (A/Bm, ~4 MB) resident in the 192 MB L2.
        f32x4 a0 = __builtin_nontemporal_load((const f32x4*)(xrow + k0 + 0));
        f32x4 a1 = __builtin_nontemporal_load((const f32x4*)(xrow + k0 + 4));
        f32x4 a2 = __builtin_nontemporal_load((const f32x4*)(xrow + k0 + 16));
        f32x4 a3 = __builtin_nontemporal_load((const f32x4*)(xrow + k0 + 20));
        // Bm is re-read by ~128 workgroups -> default (RT) caching.
        f32x4 b0 = *(const f32x4*)(brow + k0 + 0);
        f32x4 b1 = *(const f32x4*)(brow + k0 + 4);
        f32x4 b2 = *(const f32x4*)(brow + k0 + 16);
        f32x4 b3 = *(const f32x4*)(brow + k0 + 20);

        v16h af, bf;
#pragma unroll
        for (int e = 0; e < 4; ++e) {
            af[e]      = (_Float16)a0[e];
            af[4 + e]  = (_Float16)a1[e];
            af[8 + e]  = (_Float16)a2[e];
            af[12 + e] = (_Float16)a3[e];
            bf[e]      = (_Float16)b0[e];
            bf[4 + e]  = (_Float16)b1[e];
            bf[8 + e]  = (_Float16)b2[e];
            bf[12 + e] = (_Float16)b3[e];
        }
        c = __builtin_amdgcn_wmma_f32_16x16x32_f16(false, af, false, bf,
                                                   (short)0, c, false, false);
    }

    // C/D layout: VGPR j of lane -> (M = j + 8*khalf, N = mrow)
#pragma unroll
    for (int j = 0; j < 8; ++j)
        hpart[wave][(j + 8 * khalf) * 16 + mrow] = c[j];
    __syncthreads();

    // Reduce the 8 partial tiles; 256 threads, one h element each.
    {
        float s = 0.f;
#pragma unroll
        for (int w = 0; w < WAVES; ++w) s += hpart[w][tid];
        hfinal[tid] = s * SCALING;
    }
    __syncthreads();

    // ---------------- Stage 2: out_tile = h(16x16) @ A[a]^T ----------------
    // h is the f16 A-frag with K = r in [0,16); K[16,32) padded with zeros.
    v16h af2;
#pragma unroll
    for (int e = 0; e < 8; ++e) {
        af2[e]     = khalf == 0 ? (_Float16)hfinal[mrow * 16 + e] : (_Float16)0.f;
        af2[8 + e] = khalf == 0 ? (_Float16)hfinal[mrow * 16 + 8 + e] : (_Float16)0.f;   // lanes 16-31 <-> K >= 16 : zero pad
    }

    const v8f zeroc = {0.f, 0.f, 0.f, 0.f, 0.f, 0.f, 0.f, 0.f};

    for (int t = 0; t < NT_PER_WAVE; ++t) {
        const int o0 = (wave * NT_PER_WAVE + t) * 16;

        // B-frag: col N = o0+mrow, K-run = r[0,16) for lanes 0-15;
        // lanes 16-31 hold K[16,32) which is the zero padding.
        v16h bf2;
        if (khalf == 0) {
            const float* arow = A + ((size_t)a * D_OUT + o0 + mrow) * RANK;
            f32x4 r0 = *(const f32x4*)(arow + 0);
            f32x4 r1 = *(const f32x4*)(arow + 4);
            f32x4 r2 = *(const f32x4*)(arow + 8);
            f32x4 r3 = *(const f32x4*)(arow + 12);
#pragma unroll
            for (int e = 0; e < 4; ++e) {
                bf2[e]      = (_Float16)r0[e];
                bf2[4 + e]  = (_Float16)r1[e];
                bf2[8 + e]  = (_Float16)r2[e];
                bf2[12 + e] = (_Float16)r3[e];
            }
        } else {
#pragma unroll
            for (int e = 0; e < 16; ++e) bf2[e] = (_Float16)0.f;
        }

        v8f d = __builtin_amdgcn_wmma_f32_16x16x32_f16(false, af2, false, bf2,
                                                       (short)0, zeroc, false, false);

        // D layout: VGPR j -> row s0 + j + 8*khalf, col o0 + mrow.
        // Output is written once and never re-read -> non-temporal stores.
        float* orow = out + (size_t)(b * SEQ + s0 + 8 * khalf) * D_OUT + o0 + mrow;
#pragma unroll
        for (int j = 0; j < 8; ++j)
            __builtin_nontemporal_store(d[j], orow + (size_t)j * D_OUT);
    }
}

extern "C" void kernel_launch(void* const* d_in, const int* in_sizes, int n_in,
                              void* d_out, int out_size, void* d_ws, size_t ws_size,
                              hipStream_t stream) {
    const float* x   = (const float*)d_in[0];
    const float* A   = (const float*)d_in[1];
    const float* Bm  = (const float*)d_in[2];
    const int*   ids = (const int*)d_in[3];
    float* out = (float*)d_out;

    (void)in_sizes; (void)n_in; (void)out_size; (void)d_ws; (void)ws_size;

    dim3 grid(B_SZ * (SEQ / 16));   // 1024 workgroups
    dim3 block(THREADS);            // 8 wave32s
    multilora_fused_kernel<<<grid, block, 0, stream>>>(x, A, Bm, ids, out);
}
